// SlidingWindowAttention_6992206758381
// MI455X (gfx1250) — matrix operand order repaired and checked by reading, not compiled, not run on hardware
//
#include <hip/hip_runtime.h>
#include <cstdint>

typedef __bf16 bf16_t;
typedef __attribute__((ext_vector_type(16))) __bf16 v16bf;
typedef __attribute__((ext_vector_type(8)))  float  v8f;
typedef __attribute__((ext_vector_type(4)))  unsigned int v4u;
typedef __attribute__((ext_vector_type(8)))  int v8i;
typedef __attribute__((ext_vector_type(4)))  int v4i;

union Frag16 { v16bf v; v4u u[2]; };

#define WMMA_BF16(a, b, c) \
  __builtin_amdgcn_wmma_f32_16x16x32_bf16(false, (a), false, (b), (short)0, (c), false, false)

constexpr int kB  = 2;
constexpr int kS  = 2048;
constexpr int kD  = 1024;
constexpr int kNH = 16;
constexpr int kHD = 64;
constexpr int kT  = kB * kS;       // 4096 tokens
constexpr float kScale = 0.125f;   // 1/sqrt(64)
constexpr float kLn1e4 = 9.210340371976184f;

// ---- LDS raw-offset helper (addrspace(3) cast -> 32-bit LDS byte offset)
typedef __attribute__((address_space(3))) const void lds_cv;
__device__ __forceinline__ unsigned lds_off(const void* p) {
  return (unsigned)(unsigned long long)(lds_cv*)p;
}
// ---- async copy global -> LDS (16 bytes), tracked by ASYNCcnt
__device__ __forceinline__ void async_b128(const void* gaddr, const void* lptr) {
  asm volatile("global_load_async_to_lds_b128 %0, %1, off"
               :: "v"(lds_off(lptr)), "v"(gaddr) : "memory");
}
#define WAIT_ASYNC(n) asm volatile("s_wait_asynccnt " #n ::: "memory")
#define WAIT_DS0()    asm volatile("s_wait_dscnt 0" ::: "memory")

// ---- Tensor Data Mover: 2D tile (16-bit elements) global -> LDS, TENSORcnt.
// D# bit layout per CDNA5 ISA 8.3-8.6. All args wave-uniform.
// This toolchain exposes the 6-arg builtin: (g0, g1, g2, g3, g4, cpol).
__device__ __forceinline__ void tdm_load_2d(const void* gaddr, unsigned lds_byte,
                                            unsigned tensor_d0, unsigned tensor_d1,
                                            unsigned stride0,   // elements between rows
                                            unsigned tile_d0, unsigned tile_d1) {
  const unsigned long long ga = (unsigned long long)gaddr;
  v4u g0;
  g0[0] = 1u;                                            // count=1, user descriptor
  g0[1] = lds_byte;                                      // lds_addr
  g0[2] = (unsigned)(ga & 0xffffffffu);                  // global_addr[31:0]
  g0[3] = (unsigned)((ga >> 32) & 0x01ffffffu)           // global_addr[56:32]
        | (2u << 30);                                    // type = 2 ("image")
  v8i g1;
  g1[0] = (int)(1u << 16);                               // data_size = 1 (2 bytes)
  g1[1] = (int)(tensor_d0 << 16);                        // tensor_dim0[15:0]
  g1[2] = (int)((tensor_d0 >> 16) | (tensor_d1 << 16));  // dim0[31:16] | dim1[15:0]
  g1[3] = (int)((tensor_d1 >> 16) | (tile_d0 << 16));    // dim1[31:16] | tile_dim0
  g1[4] = (int)(tile_d1 & 0xffffu);                      // tile_dim1 ; tile_dim2=0
  g1[5] = (int)stride0;                                  // tensor_dim0_stride[31:0]
  g1[6] = 0;                                             // stride0[47:32], stride1[15:0]
  g1[7] = 0;                                             // stride1[47:16]
  const v4i z4 = {};
  const v8i z8 = {};
  __builtin_amdgcn_tensor_load_to_lds(g0, g1, z4, z4, z8, 0);
}

// ---------------------------------------------------------------- convert
__global__ __launch_bounds__(256) void f32_to_bf16_kernel(const float* __restrict__ in,
                                                          bf16_t* __restrict__ out, int n) {
  int i = blockIdx.x * 256 + threadIdx.x;
  if (i < n) out[i] = (__bf16)in[i];
}

// ---------------------------------------------------------------- GEMM (+RoPE / transpose / f32-out)
// C = A[T,1024] @ W[1024,1024]^T.  Block: 8 waves; each wave computes a
// 32x64 tile (2 A-frags share 4 B-frags -> 1 global b128 per WMMA). The
// 32-row A K-slab (32x32 bf16 = 2KB) is async-staged into LDS (double
// buffered) and shared by all 8 waves.
// mode 0: RoPE -> outb [B,NH,S,HD]   (Q, K)
// mode 1: transpose -> outb [B,NH,HD,S]  (V)
// mode 2: f32 -> outf [T, 1024]      (final O @ Wo^T)
__global__ __launch_bounds__(256) void gemm_bf16_kernel(const bf16_t* __restrict__ A,
                                                        const bf16_t* __restrict__ W,
                                                        bf16_t* __restrict__ outb,
                                                        float* __restrict__ outf,
                                                        int mode) {
  __shared__ bf16_t abuf[2][32 * 32];   // double-buffered A K-slab (2KB each)
  __shared__ float  lds[8][16 * 64];    // epilogue staging (per-wave)

  const int tid  = threadIdx.x;
  const int lane = tid & 31;
  const int wid  = tid >> 5;
  const int hi   = lane >> 4;
  const int col  = lane & 15;

  const int t0 = blockIdx.x * 32;               // row tile (tokens)
  const int cg = blockIdx.y * 8 + wid;          // column group 0..15 (== head for modes 0/1)
  const int j0 = cg * 64;

  // async staging assignment: 32 rows x 4 chunks of 16B (threads 0..127)
  const int arow = (tid & 127) >> 2;
  const int ach  = tid & 3;
  if (tid < 128)
    async_b128(A + (size_t)(t0 + arow) * kD + ach * 8, &abuf[0][arow * 32 + ach * 8]);

  v8f c[2][4] = {};

  for (int i = 0; i < 32; i++) {            // K-steps of 32
    __syncthreads();                        // prior reads of abuf[(i+1)&1] done
    if (i + 1 < 32) {
      if (tid < 128)
        async_b128(A + (size_t)(t0 + arow) * kD + (i + 1) * 32 + ach * 8,
                   &abuf[(i + 1) & 1][arow * 32 + ach * 8]);
      WAIT_ASYNC(0x1);                      // current slab done (in-order completion)
    } else {
      WAIT_ASYNC(0x0);
    }
    __syncthreads();                        // slab visible to all waves

    Frag16 fa0, fa1;
    const bf16_t* pa0 = &abuf[i & 1][col * 32 + hi * 8];
    fa0.u[0] = *(const v4u*)pa0;
    fa0.u[1] = *(const v4u*)(pa0 + 16);
    const bf16_t* pa1 = &abuf[i & 1][(16 + col) * 32 + hi * 8];
    fa1.u[0] = *(const v4u*)pa1;
    fa1.u[1] = *(const v4u*)(pa1 + 16);
#pragma unroll
    for (int t = 0; t < 4; t++) {
      Frag16 fb;
      const bf16_t* pb = W + (size_t)(j0 + t * 16 + col) * kD + i * 32 + hi * 8;
      fb.u[0] = *(const v4u*)pb;
      fb.u[1] = *(const v4u*)(pb + 16);
      c[0][t] = WMMA_BF16(fa0.v, fb.v, c[0][t]);
      c[1][t] = WMMA_BF16(fa1.v, fb.v, c[1][t]);
    }
  }

  if (mode == 2) {
#pragma unroll
    for (int rh = 0; rh < 2; rh++)
#pragma unroll
      for (int t = 0; t < 4; t++)
#pragma unroll
        for (int i = 0; i < 8; i++)
          outf[(size_t)(t0 + rh * 16 + i + hi * 8) * kD + j0 + t * 16 + col] = c[rh][t][i];
    return;
  }

  // per-wave epilogue (wave-local LDS staging; DS in-order + fences suffice)
  float* L = lds[wid];
  const int h = cg;  // head
  for (int rh = 0; rh < 2; rh++) {
#pragma unroll
    for (int t = 0; t < 4; t++)
#pragma unroll
      for (int i = 0; i < 8; i++)
        L[(i + hi * 8) * 64 + t * 16 + col] = c[rh][t][i];
    WAIT_DS0();
    for (int e = lane; e < 16 * 64; e += 32) {
      const int m = e >> 6;
      const int d = e & 63;
      const int tok = t0 + rh * 16 + m;
      const int b = tok >> 11;
      const int s = tok & (kS - 1);
      const float val = L[m * 64 + d];
      if (mode == 0) {  // RoPE
        const int d2 = d & 31;
        const float inv_freq = __expf(-(float)(2 * d2) * (1.0f / 64.0f) * kLn1e4);
        const float ang = (float)s * inv_freq;
        float sn, cs;
        __sincosf(ang, &sn, &cs);
        const float other = (d < 32) ? -L[m * 64 + d + 32] : L[m * 64 + d - 32];
        const float res = val * cs + other * sn;
        outb[((size_t)((b * kNH + h) * kS + s)) * kHD + d] = (__bf16)res;
      } else {          // V transposed: [B,NH,HD,S]
        outb[((size_t)((b * kNH + h) * kHD + d)) * kS + s] = (__bf16)val;
      }
    }
    WAIT_DS0();         // reads done before next pass overwrites
  }
}

// ---------------------------------------------------------------- flash attention
// Block = 8 waves = 8 consecutive query tiles of ONE (b,h). K/V^T tiles for
// each 32-key block are DMA'd into LDS by the Tensor Data Mover (one
// TENSOR_LOAD_TO_LDS per tile, issued by wave 0, double buffered,
// synchronized with s_wait_tensorcnt + block barriers).
__global__ __launch_bounds__(256) void attn_kernel(const bf16_t* __restrict__ Q,
                                                   const bf16_t* __restrict__ K,
                                                   const bf16_t* __restrict__ VT,
                                                   bf16_t* __restrict__ O) {
  __shared__ bf16_t kbuf[2][32 * 64];   // [key][dim]   4KB each
  __shared__ bf16_t vbuf[2][64 * 32];   // [dim][key]   4KB each
  __shared__ bf16_t plds[8][16 * 32];   // per-wave P transpose staging

  const int tid  = threadIdx.x;
  const int lane = tid & 31;
  const int wid  = tid >> 5;
  const int hi   = lane >> 4;
  const int col  = lane & 15;

  const int bh = blockIdx.x >> 4;        // 0..31 (b*16 + h)
  const int qg = blockIdx.x & 15;        // query-tile group
  const int qt = qg * 8 + wid;           // this wave's query tile (0..127)
  const int b  = bh >> 4;
  const int h  = bh & 15;

  const bf16_t* Qh = Q  + (size_t)bh * kS * kHD;
  const bf16_t* Kh = K  + (size_t)bh * kS * kHD;
  const bf16_t* Vh = VT + (size_t)bh * kHD * kS;

  const int my_nkb  = ((qt + 1) * 16 + 31) >> 5;  // causal trip count for this wave
  const int nkb_max = (qg + 1) * 4;               // block-max trip count

  // prologue: TDM-stage key-block 0 (wave 0 only; EXEC-independent wave op)
  if (wid == 0) {
    tdm_load_2d(Kh, lds_off(&kbuf[0][0]), kHD, kS, kHD, kHD, 32);       // 32x64 tile
    tdm_load_2d(Vh, lds_off(&vbuf[0][0]), kS, kHD, kS, 32, kHD);        // 64x32 tile
  }

  // preload Q A-fragments (16 rows x 64 head-dims = 2 frags)
  Frag16 qf[2];
#pragma unroll
  for (int kk = 0; kk < 2; kk++) {
    const bf16_t* p = Qh + (size_t)(qt * 16 + col) * kHD + kk * 32 + hi * 8;
    qf[kk].u[0] = *(const v4u*)p;
    qf[kk].u[1] = *(const v4u*)(p + 16);
  }

  v8f o[4] = {};
  float m_i[8], l_i[8];
#pragma unroll
  for (int i = 0; i < 8; i++) { m_i[i] = -1e30f; l_i[i] = 0.0f; }

  for (int kb = 0; kb < nkb_max; kb++) {
    __syncthreads();                         // prior reads of next buffer done
    if (wid == 0) {
      if (kb + 1 < nkb_max) {
        const int ks1 = (kb + 1) * 32;
        const int nb  = (kb + 1) & 1;
        tdm_load_2d(Kh + (size_t)ks1 * kHD, lds_off(&kbuf[nb][0]), kHD, kS, kHD, kHD, 32);
        tdm_load_2d(Vh + ks1,               lds_off(&vbuf[nb][0]), kS, kHD, kS, 32, kHD);
        __builtin_amdgcn_s_wait_tensorcnt((short)2);  // current tile done (in-order)
      } else {
        __builtin_amdgcn_s_wait_tensorcnt((short)0);
      }
    }
    __syncthreads();                         // tile visible to all waves

    if (kb < my_nkb) {
      const int ks = kb * 32;
      const bf16_t* kb_c = kbuf[kb & 1];
      const bf16_t* vb_c = vbuf[kb & 1];

      // ---- scores: Q [16x64] @ K^T [64x32] -> two 16x16 f32 tiles
      v8f s0 = {}, s1 = {};
#pragma unroll
      for (int kk = 0; kk < 2; kk++) {
        Frag16 b0, b1;
        const bf16_t* pk0 = kb_c + (size_t)col * kHD + kk * 32 + hi * 8;
        b0.u[0] = *(const v4u*)pk0;
        b0.u[1] = *(const v4u*)(pk0 + 16);
        const bf16_t* pk1 = kb_c + (size_t)(16 + col) * kHD + kk * 32 + hi * 8;
        b1.u[0] = *(const v4u*)pk1;
        b1.u[1] = *(const v4u*)(pk1 + 16);
        s0 = WMMA_BF16(qf[kk].v, b0.v, s0);
        s1 = WMMA_BF16(qf[kk].v, b1.v, s1);
      }

      const bool last = (kb == my_nkb - 1);
#pragma unroll
      for (int i = 0; i < 8; i++) { s0[i] *= kScale; s1[i] *= kScale; }
      if (last) {
#pragma unroll
        for (int i = 0; i < 8; i++) {
          const int qpos = qt * 16 + i + hi * 8;
          if (ks + col > qpos)      s0[i] = -1e30f;
          if (ks + 16 + col > qpos) s1[i] = -1e30f;
        }
      }

      // ---- online softmax (row reductions within 16-lane halves)
#pragma unroll
      for (int i = 0; i < 8; i++) {
        float mx = fmaxf(s0[i], s1[i]);
        mx = fmaxf(mx, __shfl_xor(mx, 1, 32));
        mx = fmaxf(mx, __shfl_xor(mx, 2, 32));
        mx = fmaxf(mx, __shfl_xor(mx, 4, 32));
        mx = fmaxf(mx, __shfl_xor(mx, 8, 32));
        const float mnew = fmaxf(m_i[i], mx);
        const float p0 = __expf(s0[i] - mnew);
        const float p1 = __expf(s1[i] - mnew);
        const float corr = __expf(m_i[i] - mnew);
        m_i[i] = mnew;
        float rs = p0 + p1;
        rs += __shfl_xor(rs, 1, 32);
        rs += __shfl_xor(rs, 2, 32);
        rs += __shfl_xor(rs, 4, 32);
        rs += __shfl_xor(rs, 8, 32);
        l_i[i] = l_i[i] * corr + rs;
        o[0][i] *= corr; o[1][i] *= corr; o[2][i] *= corr; o[3][i] *= corr;
        s0[i] = p0; s1[i] = p1;
      }

      // ---- transpose P (C layout) -> A layout through per-wave LDS
      bf16_t* pl = plds[wid];
#pragma unroll
      for (int i = 0; i < 8; i++) {
        const int M = i + hi * 8;
        pl[M * 32 + col]      = (__bf16)s0[i];
        pl[M * 32 + 16 + col] = (__bf16)s1[i];
      }
      WAIT_DS0();
      Frag16 pa;
      {
        const bf16_t* pp = pl + col * 32 + hi * 8;
        pa.u[0] = *(const v4u*)pp;
        pa.u[1] = *(const v4u*)(pp + 16);
      }

      // ---- O += P [16x32] @ V [32x64]
#pragma unroll
      for (int t = 0; t < 4; t++) {
        Frag16 bv;
        const bf16_t* pv = vb_c + (size_t)(t * 16 + col) * 32 + hi * 8;
        bv.u[0] = *(const v4u*)pv;
        bv.u[1] = *(const v4u*)(pv + 16);
        o[t] = WMMA_BF16(pa.v, bv.v, o[t]);
      }
    }
  }

  // ---- normalize and store to O buffer [T, D] (token-major for final GEMM)
#pragma unroll
  for (int t = 0; t < 4; t++) {
#pragma unroll
    for (int i = 0; i < 8; i++) {
      const int M = i + hi * 8;
      const size_t idx = (size_t)(b * kS + qt * 16 + M) * kD + h * 64 + t * 16 + col;
      O[idx] = (__bf16)(o[t][i] * (1.0f / l_i[i]));
    }
  }
}

// ---------------------------------------------------------------- launcher
extern "C" void kernel_launch(void* const* d_in, const int* in_sizes, int n_in,
                              void* d_out, int out_size, void* d_ws, size_t ws_size,
                              hipStream_t stream) {
  const float* x  = (const float*)d_in[0];
  const float* Wq = (const float*)d_in[1];
  const float* Wk = (const float*)d_in[2];
  const float* Wv = (const float*)d_in[3];
  const float* Wo = (const float*)d_in[4];
  float* out = (float*)d_out;

  char* ws = (char*)d_ws;
  const size_t MB = 1u << 20;
  bf16_t* xb  = (bf16_t*)(ws + 0 * MB);   // [T, D]        8 MB
  bf16_t* Wqb = (bf16_t*)(ws + 8 * MB);   //               2 MB
  bf16_t* Wkb = (bf16_t*)(ws + 10 * MB);
  bf16_t* Wvb = (bf16_t*)(ws + 12 * MB);
  bf16_t* Wob = (bf16_t*)(ws + 14 * MB);
  bf16_t* Qb  = (bf16_t*)(ws + 16 * MB);  // [B,NH,S,HD]   8 MB
  bf16_t* Kb  = (bf16_t*)(ws + 24 * MB);  // [B,NH,S,HD]   8 MB
  bf16_t* Vt  = (bf16_t*)(ws + 32 * MB);  // [B,NH,HD,S]   8 MB
  bf16_t* Ob  = (bf16_t*)(ws + 40 * MB);  // [T, D]        8 MB

  const int nX = kT * kD;
  const int nW = kD * kD;

  f32_to_bf16_kernel<<<nX / 256, 256, 0, stream>>>(x, xb, nX);
  f32_to_bf16_kernel<<<nW / 256, 256, 0, stream>>>(Wq, Wqb, nW);
  f32_to_bf16_kernel<<<nW / 256, 256, 0, stream>>>(Wk, Wkb, nW);
  f32_to_bf16_kernel<<<nW / 256, 256, 0, stream>>>(Wv, Wvb, nW);
  f32_to_bf16_kernel<<<nW / 256, 256, 0, stream>>>(Wo, Wob, nW);

  dim3 ggrid(kT / 32, 2);  // 128 row tiles x (16 col groups / 8 waves)
  gemm_bf16_kernel<<<ggrid, 256, 0, stream>>>(xb, Wqb, Qb, nullptr, 0);  // Q + RoPE
  gemm_bf16_kernel<<<ggrid, 256, 0, stream>>>(xb, Wkb, Kb, nullptr, 0);  // K + RoPE
  gemm_bf16_kernel<<<ggrid, 256, 0, stream>>>(xb, Wvb, Vt, nullptr, 1);  // V transposed

  attn_kernel<<<(kB * kNH) * 16, 256, 0, stream>>>(Qb, Kb, Vt, Ob);

  gemm_bf16_kernel<<<ggrid, 256, 0, stream>>>(Ob, Wob, nullptr, out, 2); // O @ Wo^T -> f32
}
